// GRU_69406671504098
// MI455X (gfx1250) — matrix operand order repaired and checked by reading, not compiled, not run on hardware
//
#include <hip/hip_runtime.h>

// ---------------------------------------------------------------------------
// GRU forward, MI455X (gfx1250, wave32, WMMA).
// T=32, B=128, N=128, D=2, H=128.  Rows = B*N = 16384.
//
//  - Input path folded to rank-2: gi = inputs @ Weff^T + beff,
//    Weff = W_ih @ W1 (384x2), beff = W_ih @ b1 + b_ih.
//  - Recurrent GEMM per step: gh = h @ W_hh^T, M=16384,N=384,K=128 via
//    v_wmma_f32_16x16x32_f16 (f16 operands, fp32 accum).
//  - 64-row blocks: B fragments (W_hh columns) cached in VGPRs and reused
//    across 4 row tiles -> 4x less L2 traffic for W_hh.
//  - Elementwise gates use v_exp2/v_rcp fast paths (no precise division),
//    biases folded, update as a single fma: VALU is the secondary roofline.
//  - h_new[row] depends only on h_old[row] -> in-place update, no inter-block
//    sync inside a step; 32 sequential kernel launches order the recurrence.
// ---------------------------------------------------------------------------

typedef __attribute__((ext_vector_type(16))) _Float16 v16h;
typedef __attribute__((ext_vector_type(8)))  _Float16 v8h;
typedef __attribute__((ext_vector_type(8)))  float    v8f;

constexpr int T  = 32;
constexpr int Bb = 128;
constexpr int Nn = 128;
constexpr int Dd = 2;
constexpr int H  = 128;
constexpr int BN = Bb * Nn;     // 16384 rows
constexpr int G3 = 3 * H;       // 384 gate rows
constexpr int MROWS = 64;       // rows per block (4 row tiles of 16)

union V16H { v16h v; v8h h[2]; };

// sigmoid(x) = rcp(1 + exp2(-log2(e) * x))          : 4 VALU ops
__device__ __forceinline__ float fast_sigmoid(float x) {
    const float e = __builtin_amdgcn_exp2f(x * -1.442695040888963f);
    return __builtin_amdgcn_rcpf(1.0f + e);
}
// tanh(x) = 2*sigmoid(2x) - 1                       : 5 VALU ops
__device__ __forceinline__ float fast_tanh(float x) {
    const float e = __builtin_amdgcn_exp2f(x * -2.885390081777927f);
    const float s = __builtin_amdgcn_rcpf(1.0f + e);
    return __builtin_fmaf(2.0f, s, -1.0f);
}

// ---------------------------------------------------------------------------
// Prep: Weff (384x2), beff (384), W_hh -> f16 copy.
// One block per gate row g (384 blocks x 128 threads).
// ---------------------------------------------------------------------------
__global__ void __launch_bounds__(128)
gru_prep(const float* __restrict__ W1, const float* __restrict__ b1,
         const float* __restrict__ W_ih, const float* __restrict__ b_ih,
         const float* __restrict__ W_hh,
         float* __restrict__ Weff, float* __restrict__ beff,
         _Float16* __restrict__ Whh_h) {
    const int g = blockIdx.x;       // 0..383
    const int t = threadIdx.x;      // 0..127
    const float wih = W_ih[g * H + t];
    Whh_h[g * H + t] = (_Float16)W_hh[g * H + t];

    __shared__ float s0[128], s1[128], s2[128];
    s0[t] = wih * W1[t * Dd + 0];
    s1[t] = wih * W1[t * Dd + 1];
    s2[t] = wih * b1[t];
    __syncthreads();
    for (int s = 64; s > 0; s >>= 1) {
        if (t < s) { s0[t] += s0[t + s]; s1[t] += s1[t + s]; s2[t] += s2[t + s]; }
        __syncthreads();
    }
    if (t == 0) {
        Weff[g * 2 + 0] = s0[0];
        Weff[g * 2 + 1] = s1[0];
        beff[g]         = s2[0] + b_ih[g];
    }
}

// ---------------------------------------------------------------------------
// One GRU timestep. Block = 64 rows x all 128 H-cols, 8 waves.
// Wave w owns H-cols [16w,16w+16); B fragments cached in VGPRs, reused over
// 4 row tiles of 16. Per wave: 4 tiles x 3 gates x 4 kb = 48 WMMAs.
// ---------------------------------------------------------------------------
__global__ void __launch_bounds__(256)
gru_step(const float* __restrict__ inputs_t,     // (BN, D) slice for this t
         const _Float16* __restrict__ Whh_h,     // (3H, H) f16, row-major
         const float* __restrict__ Weff,         // (3H, 2)
         const float* __restrict__ beff,         // (3H)
         const float* __restrict__ b_hh,         // (3H)
         float* __restrict__ hbuf) {             // (BN, H), updated in place
    __shared__ float    lds_hf[MROWS * H];       // fp32 h tile (for update)
    __shared__ _Float16 lds_h16[MROWS * H];      // f16 h tile (WMMA A)
    __shared__ float    lds_in[MROWS * Dd];      // input tile

    const int tid  = threadIdx.x;
    const int wave = tid >> 5;
    const int lane = tid & 31;
    const int row0 = blockIdx.x * MROWS;

    for (int i = tid; i < MROWS * H; i += 256) {
        const float v = hbuf[(size_t)row0 * H + i];
        lds_hf[i]  = v;
        lds_h16[i] = (_Float16)v;
    }
    if (tid < MROWS * Dd) lds_in[tid] = inputs_t[(size_t)row0 * Dd + tid];
    __syncthreads();

    const int l15 = lane & 15;          // A: row M  | B/C/D: col N
    const int hi  = lane >> 4;          // lane half selects K/M offsets
    const int gh_col = wave * 16 + l15; // H column this lane produces
    const int g_r = gh_col, g_z = H + gh_col, g_n = 2 * H + gh_col;

    // ---- Cache B fragments for all 4 K-blocks x 3 gates in VGPRs ----------
    // B 32x16 f16 operand, this code's k order (same as A): per lane (col g), 16 contiguous K at k0 + 16*hi.
    V16H Br[4], Bz[4], Bn[4];
    #pragma unroll
    for (int kb = 0; kb < 4; ++kb) {
        const _Float16* bp = Whh_h + kb * 32 + hi * 16;
        Br[kb].v = *(const v16h*)(bp + (size_t)g_r * H);
        Bz[kb].v = *(const v16h*)(bp + (size_t)g_z * H);
        Bn[kb].v = *(const v16h*)(bp + (size_t)g_n * H);
    }

    // ---- Per-lane gate constants (g fixed per lane) -----------------------
    // r/z biases fold beff + b_hh; n-gate keeps b_hh separate (scaled by r).
    const float wr0 = Weff[g_r * 2], wr1 = Weff[g_r * 2 + 1];
    const float wz0 = Weff[g_z * 2], wz1 = Weff[g_z * 2 + 1];
    const float wn0 = Weff[g_n * 2], wn1 = Weff[g_n * 2 + 1];
    const float bcr = beff[g_r] + b_hh[g_r];
    const float bcz = beff[g_z] + b_hh[g_z];
    const float ben = beff[g_n];
    const float bhn = b_hh[g_n];

    // ---- 4 row tiles of 16, reusing the cached B fragments ----------------
    #pragma unroll
    for (int rt = 0; rt < 4; ++rt) {
        const int rbase = rt * 16;
        v8f acc_r = {}, acc_z = {}, acc_n = {};
        #pragma unroll
        for (int kb = 0; kb < 4; ++kb) {
            const int k0 = kb * 32;
            // A 16x32 f16 operand, this code's k order (same as B): per lane, 16 contiguous K at k0 + 16*hi
            V16H a;
            a.h[0] = *(const v8h*)(lds_h16 + (rbase + l15) * H + k0 + hi * 16);
            a.h[1] = *(const v8h*)(lds_h16 + (rbase + l15) * H + k0 + hi * 16 + 8);
            acc_r = __builtin_amdgcn_wmma_f32_16x16x32_f16(false, a.v, false, Br[kb].v,
                                                           (short)0, acc_r, false, false);
            acc_z = __builtin_amdgcn_wmma_f32_16x16x32_f16(false, a.v, false, Bz[kb].v,
                                                           (short)0, acc_z, false, false);
            acc_n = __builtin_amdgcn_wmma_f32_16x16x32_f16(false, a.v, false, Bn[kb].v,
                                                           (short)0, acc_n, false, false);
        }

        #pragma unroll
        for (int p = 0; p < 8; ++p) {
            const int rl = rbase + p + 8 * hi;        // local row in block
            const float in0 = lds_in[rl * 2 + 0];
            const float in1 = lds_in[rl * 2 + 1];
            const float pre_r = __builtin_fmaf(in1, wr1, __builtin_fmaf(in0, wr0, bcr)) + acc_r[p];
            const float pre_z = __builtin_fmaf(in1, wz1, __builtin_fmaf(in0, wz0, bcz)) + acc_z[p];
            const float gi_n  = __builtin_fmaf(in1, wn1, __builtin_fmaf(in0, wn0, ben));
            const float ghn   = acc_n[p] + bhn;
            const float r = fast_sigmoid(pre_r);
            const float z = fast_sigmoid(pre_z);
            const float n = fast_tanh(__builtin_fmaf(r, ghn, gi_n));
            const float hold = lds_hf[rl * H + gh_col];
            // (1-z)*n + z*hold == n + z*(hold - n)
            hbuf[(size_t)(row0 + rl) * H + gh_col] = __builtin_fmaf(z, hold - n, n);
        }
    }
}

// ---------------------------------------------------------------------------
// Output projection: pred = h @ Wo^T + bo.  One wave per row.
// ---------------------------------------------------------------------------
__global__ void __launch_bounds__(256)
gru_out(const float* __restrict__ hbuf, const float* __restrict__ Wo,
        const float* __restrict__ bo, float* __restrict__ pred) {
    const int wave = threadIdx.x >> 5, lane = threadIdx.x & 31;
    const int row = blockIdx.x * 8 + wave;
    const float* hr = hbuf + (size_t)row * H;
    float s0 = 0.f, s1 = 0.f;
    #pragma unroll
    for (int k = lane; k < H; k += 32) {
        const float hv = hr[k];
        s0 += hv * Wo[0 * H + k];
        s1 += hv * Wo[1 * H + k];
    }
    #pragma unroll
    for (int m = 16; m; m >>= 1) {
        s0 += __shfl_xor(s0, m, 32);
        s1 += __shfl_xor(s1, m, 32);
    }
    if (lane == 0) {
        pred[row * 2 + 0] = s0 + bo[0];
        pred[row * 2 + 1] = s1 + bo[1];
    }
}

// ---------------------------------------------------------------------------
extern "C" void kernel_launch(void* const* d_in, const int* in_sizes, int n_in,
                              void* d_out, int out_size, void* d_ws, size_t ws_size,
                              hipStream_t stream) {
    const float* inputs = (const float*)d_in[0];   // (T,B,N,D)
    const float* hidden = (const float*)d_in[1];   // (B,N,H)
    const float* W1     = (const float*)d_in[2];   // (H,D)
    const float* b1     = (const float*)d_in[3];   // (H)
    const float* W_ih   = (const float*)d_in[4];   // (3H,H)
    const float* W_hh   = (const float*)d_in[5];   // (3H,H)
    const float* b_ih   = (const float*)d_in[6];   // (3H)
    const float* b_hh   = (const float*)d_in[7];   // (3H)
    const float* Wo     = (const float*)d_in[8];   // (D,H)
    const float* bo     = (const float*)d_in[9];   // (D)
    float* pred = (float*)d_out;                   // (B,N,D)

    char* ws = (char*)d_ws;
    float*    hbuf  = (float*)ws;     ws += (size_t)BN * H * sizeof(float);      // 8 MB
    _Float16* Whh_h = (_Float16*)ws;  ws += (size_t)G3 * H * sizeof(_Float16);   // 96 KB
    float*    Weff  = (float*)ws;     ws += (size_t)G3 * 2 * sizeof(float);
    float*    beff  = (float*)ws;     ws += (size_t)G3 * sizeof(float);
    (void)in_sizes; (void)n_in; (void)out_size; (void)ws_size;

    hipMemcpyAsync(hbuf, hidden, (size_t)BN * H * sizeof(float),
                   hipMemcpyDeviceToDevice, stream);
    gru_prep<<<G3, H, 0, stream>>>(W1, b1, W_ih, b_ih, W_hh, Weff, beff, Whh_h);

    for (int t = 0; t < T; ++t) {
        gru_step<<<BN / MROWS, 256, 0, stream>>>(inputs + (size_t)t * BN * Dd,
                                                 Whh_h, Weff, beff, b_hh, hbuf);
    }
    gru_out<<<BN / 8, 256, 0, stream>>>(hbuf, Wo, bo, pred);
}
